// HierarchicalCooccurrenceTransformer_36953898615253
// MI455X (gfx1250) — matrix operand order repaired and checked by reading, not compiled, not run on hardware
//
#include <hip/hip_runtime.h>
#include <math.h>

// ---------------- problem constants ----------------
#define BB   64
#define WW   128
#define LL   20
#define DD   128
#define FFN  256
#define NH   4
#define DH   32
static const size_t M1 = (size_t)BB * WW * LL;   // 163840 token rows (intra)
static const size_t M2 = (size_t)BB * WW;        // 8192 walk rows (inter)

typedef _Float16 h16;
typedef __attribute__((ext_vector_type(16))) _Float16 v16h;
typedef __attribute__((ext_vector_type(8)))  _Float16 v8h;
typedef __attribute__((ext_vector_type(8)))  float    v8f;
typedef __attribute__((ext_vector_type(4)))  unsigned int u32x4;
typedef __attribute__((ext_vector_type(8)))  int i32x8;
typedef __attribute__((ext_vector_type(4)))  int i32x4;

#if defined(__has_builtin)
#  if __has_builtin(__builtin_amdgcn_tensor_load_to_lds) && __has_builtin(__builtin_amdgcn_s_wait_tensorcnt)
#    define HAVE_TDM 1
#  endif
#endif

#ifdef HAVE_TDM
// Issue a 2-D TDM load: tile [tile1 rows x tile0 elts] of a row-major f16
// tensor (row stride = stride0 elements) starting at gsrc, into LDS at lds_off.
// Rows beyond tdim1 / elements beyond tdim0 are zero-filled by the TDM.
__device__ static inline void tdm_load_2d_f16(const void* gsrc, unsigned lds_off,
                                              unsigned tdim0, unsigned tdim1,
                                              unsigned tile0, unsigned tile1,
                                              unsigned long long stride0) {
    unsigned long long ga = (unsigned long long)(uintptr_t)gsrc;
    u32x4 g0;
    g0[0] = 1u;                                            // count=1 (valid user D#)
    g0[1] = lds_off;                                       // lds_addr
    g0[2] = (unsigned)(ga & 0xffffffffu);                  // global_addr[31:0]
    g0[3] = (unsigned)((ga >> 32) & 0x01ffffffu) | (2u << 30); // addr[56:32] | type=2
    i32x8 g1;
    g1[0] = (int)(1u << 16);                               // wg_mask=0, data_size=1 (2B)
    g1[1] = (int)((tdim0 & 0xffffu) << 16);                // tensor_dim0[15:0] @ bits63:48
    g1[2] = (int)((tdim0 >> 16) | ((tdim1 & 0xffffu) << 16)); // td0[31:16], td1[15:0]
    g1[3] = (int)((tdim1 >> 16) | (tile0 << 16));          // td1[31:16], tile_dim0
    g1[4] = (int)(tile1 & 0xffffu);                        // tile_dim1, tile_dim2=0
    g1[5] = (int)(stride0 & 0xffffffffu);                  // tensor_dim0_stride[31:0]
    g1[6] = (int)((stride0 >> 32) & 0xffffu);              // stride0[47:32], stride1=0
    g1[7] = 0;
    i32x4 gz4 = {0, 0, 0, 0};
    i32x8 gz8 = {0, 0, 0, 0, 0, 0, 0, 0};
    // 6-arg toolchain variant: (g0, g1, g2, g3, g4, cpol)
    __builtin_amdgcn_tensor_load_to_lds(g0, g1, gz4, gz4, gz8, 0);
}
#endif

// ---------------- f32 -> f16 weight conversion ----------------
__global__ void k_cvt(const float* __restrict__ s, h16* __restrict__ d, int n) {
    int i = blockIdx.x * 256 + threadIdx.x;
    if (i < n) d[i] = (h16)s[i];
}

// ---------------- embed: x = ne + type_embed[0] + PE(pos,d) ----------------
__global__ void k_embed(const float* __restrict__ ne, const float* __restrict__ te,
                        h16* __restrict__ x, long n) {
    long idx = (long)blockIdx.x * 256 + threadIdx.x;
    if (idx >= n) return;
    int d   = (int)(idx & (DD - 1));
    long tok = idx >> 7;              // /128
    int pos = (int)(tok % LL);
    int pair = d >> 1;
    float div = __expf((float)(2 * pair) * (-9.210340372f / (float)DD)); // -ln(10000)/D
    float pe = (d & 1) ? __cosf((float)pos * div) : __sinf((float)pos * div);
    x[idx] = (h16)(ne[idx] + te[d] + pe);
}

// ---------------- WMMA GEMM: C[M,N](f16) = act(A[M,K]f16 @ B[K,N]f16 + bias) -------
// block 256 thr = 8 waves; tile 128x64, BK=32; wave grid 4x2, each wave 32x32 (2x2 WMMA)
// A tile staged by the Tensor Data Mover (double buffered, 1 barrier / K-step).
// Requires: N % 64 == 0, K % 32 == 0 (true for all call sites).
__global__ __launch_bounds__(256)
void k_gemm(const h16* __restrict__ A, const h16* __restrict__ Bw,
            const float* __restrict__ bias, h16* __restrict__ C,
            int M, int N, int K, int act) {
    __shared__ h16 As[2][128][32];   // [buf][m][k]
    __shared__ h16 Bst[2][64][32];   // [buf][n][k]  (K-transposed)

    const int tid  = threadIdx.x;
    const int wid  = tid >> 5;
    const int lane = tid & 31;
    const int wm   = wid >> 1;          // 0..3
    const int wn   = wid & 1;           // 0..1
    const int lg   = lane >> 4;         // half-wave selector
    const int lm   = lane & 15;
    const int row0 = blockIdx.x * 128;
    const int col0 = blockIdx.y * 64;
    const unsigned rowsLeft = (unsigned)(M - row0);   // rows available (>=1)

    v8f acc[2][2] = {};

    // ---- stage B tile k0 -> buffer nb (transpose to [n][k]) ----
    auto stageB = [&](int k0, int nb) {
        int kk = tid >> 3;           // 0..31
        int n8 = (tid & 7) * 8;      // 0..56
        v8h vv = *(const v8h*)(Bw + (size_t)(k0 + kk) * N + col0 + n8);
        #pragma unroll
        for (int i = 0; i < 8; i++) Bst[nb][n8 + i][kk] = vv[i];
    };
    // ---- stage A tile k0 -> buffer nb ----
    auto stageA = [&](int k0, int nb) {
#ifdef HAVE_TDM
        if (tid < 32) {   // wave 0 issues the DMA (TDM ignores EXEC within the wave)
            tdm_load_2d_f16(A + (size_t)row0 * K + k0,
                            (unsigned)(uintptr_t)&As[nb][0][0],
                            (unsigned)(K - k0), rowsLeft, 32u, 128u,
                            (unsigned long long)K);
        }
#else
        #pragma unroll
        for (int c = tid; c < 512; c += 256) {
            int r  = c >> 2;
            int ck = (c & 3) * 8;
            int gr = row0 + r;
            v8h vv = {};
            if (gr < M) vv = *(const v8h*)(A + (size_t)gr * K + k0 + ck);
            *(v8h*)&As[nb][r][ck] = vv;
        }
#endif
    };

    // ---- prologue: fill buffer 0 ----
    stageA(0, 0);
    stageB(0, 0);
#ifdef HAVE_TDM
    __builtin_amdgcn_s_wait_tensorcnt(0);
#endif
    __syncthreads();

    int nb = 0;
    for (int k0 = 0; k0 < K; k0 += 32) {
        int nxt = k0 + 32;
        if (nxt < K) {                 // prefetch next tiles into other buffer
            stageA(nxt, nb ^ 1);
            stageB(nxt, nb ^ 1);
        }
        // ---- fragments: k = 16*half + s (contiguous) for both A and B ----
        v16h afrag[2], bfrag[2];
        #pragma unroll
        for (int i = 0; i < 2; i++) {
            int m  = wm * 32 + i * 16 + lm;
            int kb = lg * 16;
            v8h lo = *(const v8h*)&As[nb][m][kb];
            v8h hi = *(const v8h*)&As[nb][m][kb + 8];
            afrag[i] = __builtin_shufflevector(lo, hi, 0,1,2,3,4,5,6,7,8,9,10,11,12,13,14,15);
        }
        #pragma unroll
        for (int j = 0; j < 2; j++) {
            int n  = wn * 32 + j * 16 + lm;
            int kb = lg * 16;
            bfrag[j] = *(const v16h*)&Bst[nb][n][kb];
        }
        #pragma unroll
        for (int i = 0; i < 2; i++)
            #pragma unroll
            for (int j = 0; j < 2; j++)
                acc[i][j] = __builtin_amdgcn_wmma_f32_16x16x32_f16(
                    false, afrag[i], false, bfrag[j], (short)0, acc[i][j], false, false);
#ifdef HAVE_TDM
        __builtin_amdgcn_s_wait_tensorcnt(0);   // next A tile landed
#endif
        __syncthreads();                        // everyone done reading buffer nb
        nb ^= 1;
    }

    // ---- epilogue: C layout (VGPR r: lanes0-15 -> M=r, lanes16-31 -> M=r+8) ----
    #pragma unroll
    for (int i = 0; i < 2; i++) {
        #pragma unroll
        for (int j = 0; j < 2; j++) {
            int ccol = col0 + wn * 32 + j * 16 + lm;
            float bcol = bias[ccol];
            #pragma unroll
            for (int r = 0; r < 8; r++) {
                int crow = row0 + wm * 32 + i * 16 + lg * 8 + r;
                if (crow < M) {
                    float v = acc[i][j][r] + bcol;
                    if (act == 1) v = fmaxf(v, 0.f);
                    else if (act == 2) v = tanhf(v);
                    C[(size_t)crow * N + ccol] = (h16)v;
                }
            }
        }
    }
}

// ---------------- intra-walk attention: one wave per (bw, head) ----------------
__global__ __launch_bounds__(32)
void k_attn_intra(const h16* __restrict__ q, const h16* __restrict__ k,
                  const h16* __restrict__ v, const int* __restrict__ masks,
                  h16* __restrict__ o) {
    int bw = blockIdx.x >> 2;
    int h  = blockIdx.x & 3;
    int lane = threadIdx.x;
    __shared__ h16 ks[LL][DH], vs[LL][DH];
    __shared__ float mneg[LL];

    for (int idx = lane; idx < LL * DH; idx += 32) {
        int r = idx / DH, c = idx % DH;
        ks[r][c] = k[((size_t)bw * LL + r) * DD + h * DH + c];
        vs[r][c] = v[((size_t)bw * LL + r) * DD + h * DH + c];
    }
    if (lane < LL) mneg[lane] = masks[(size_t)bw * LL + lane] ? 0.f : -1e9f;
    __syncthreads();

    if (lane < LL) {
        float qv[DH];
        const h16* qp = q + ((size_t)bw * LL + lane) * DD + h * DH;
        #pragma unroll
        for (int c = 0; c < DH; c++) qv[c] = (float)qp[c];
        float sc[LL], mx = -1e30f;
        #pragma unroll
        for (int j = 0; j < LL; j++) {
            float s = 0.f;
            #pragma unroll
            for (int c = 0; c < DH; c++) s += qv[c] * (float)ks[j][c];
            s = s * 0.1767766953f + mneg[j];   // 1/sqrt(32)
            sc[j] = s; mx = fmaxf(mx, s);
        }
        float den = 0.f;
        #pragma unroll
        for (int j = 0; j < LL; j++) { sc[j] = __expf(sc[j] - mx); den += sc[j]; }
        float inv = 1.f / (den + 1e-20f);
        float ov[DH] = {};
        #pragma unroll
        for (int j = 0; j < LL; j++)
            #pragma unroll
            for (int c = 0; c < DH; c++) ov[c] += sc[j] * (float)vs[j][c];
        h16* op = o + ((size_t)bw * LL + lane) * DD + h * DH;
        #pragma unroll
        for (int c = 0; c < DH; c++) op[c] = (h16)(ov[c] * inv);
    }
}

// ---------------- inter-walk attention with cooc bias (flash-style) ----------------
__global__ __launch_bounds__(128)
void k_attn_inter(const h16* __restrict__ q, const h16* __restrict__ k,
                  const h16* __restrict__ v, const float* __restrict__ cooc,
                  const float* __restrict__ gammap, const int* __restrict__ masks,
                  h16* __restrict__ o) {
    int b = blockIdx.x >> 2;
    int h = blockIdx.x & 3;
    int t = threadIdx.x;                 // query walk
    __shared__ h16 ks[WW][DH], vs[WW][DH];
    __shared__ float kinv[WW];

    const h16* kp = k + ((size_t)b * WW + t) * DD + h * DH;
    const h16* vp = v + ((size_t)b * WW + t) * DD + h * DH;
    #pragma unroll
    for (int c = 0; c < DH; c++) { ks[t][c] = kp[c]; vs[t][c] = vp[c]; }
    int len = 0;
    const int* mp = masks + ((size_t)b * WW + t) * LL;
    for (int l = 0; l < LL; l++) len += mp[l];
    kinv[t] = len ? 0.f : -1e9f;
    __syncthreads();

    float g = gammap[0];
    float qv[DH];
    const h16* qp = q + ((size_t)b * WW + t) * DD + h * DH;
    #pragma unroll
    for (int c = 0; c < DH; c++) qv[c] = (float)qp[c];
    const float* cp = cooc + ((size_t)b * WW + t) * WW;

    float m = -1e30f, l = 0.f, ov[DH] = {};
    for (int j = 0; j < WW; j++) {
        float s = 0.f;
        #pragma unroll
        for (int c = 0; c < DH; c++) s += qv[c] * (float)ks[j][c];
        s = s * 0.1767766953f + g * cp[j] * 0.08838834765f + kinv[j]; // d^-0.5 bias
        float mn = fmaxf(m, s);
        float sc = __expf(m - mn), p = __expf(s - mn);
        l = l * sc + p;
        #pragma unroll
        for (int c = 0; c < DH; c++) ov[c] = ov[c] * sc + p * (float)vs[j][c];
        m = mn;
    }
    float inv = 1.f / (l + 1e-20f);
    h16* op = o + ((size_t)b * WW + t) * DD + h * DH;
    #pragma unroll
    for (int c = 0; c < DH; c++) op[c] = (h16)(ov[c] * inv);
}

// ---------------- LayerNorm over D=128 (one wave per row), optional residual -------
__global__ __launch_bounds__(32)
void k_ln(const h16* __restrict__ x, const h16* __restrict__ r,
          const float* __restrict__ s, const float* __restrict__ b,
          h16* __restrict__ out, int M) {
    int row = blockIdx.x;
    if (row >= M) return;
    int lane = threadIdx.x;
    size_t base = (size_t)row * DD + lane * 4;
    float v[4];
    #pragma unroll
    for (int i = 0; i < 4; i++) {
        v[i] = (float)x[base + i];
        if (r) v[i] += (float)r[base + i];
    }
    float sum = v[0] + v[1] + v[2] + v[3];
    for (int off = 16; off; off >>= 1) sum += __shfl_xor(sum, off, 32);
    float mu = sum * (1.f / DD);
    float var = 0.f;
    #pragma unroll
    for (int i = 0; i < 4; i++) { float d = v[i] - mu; var += d * d; }
    for (int off = 16; off; off >>= 1) var += __shfl_xor(var, off, 32);
    var *= (1.f / DD);
    float inv = rsqrtf(var + 1e-5f);
    #pragma unroll
    for (int i = 0; i < 4; i++)
        out[base + i] = (h16)((v[i] - mu) * inv * s[lane * 4 + i] + b[lane * 4 + i]);
}

// f32-output LN for the final result
__global__ __launch_bounds__(32)
void k_ln_f32(const h16* __restrict__ x, const float* __restrict__ s,
              const float* __restrict__ b, float* __restrict__ out) {
    int row = blockIdx.x;
    int lane = threadIdx.x;
    size_t base = (size_t)row * DD + lane * 4;
    float v[4];
    #pragma unroll
    for (int i = 0; i < 4; i++) v[i] = (float)x[base + i];
    float sum = v[0] + v[1] + v[2] + v[3];
    for (int off = 16; off; off >>= 1) sum += __shfl_xor(sum, off, 32);
    float mu = sum * (1.f / DD);
    float var = 0.f;
    #pragma unroll
    for (int i = 0; i < 4; i++) { float d = v[i] - mu; var += d * d; }
    for (int off = 16; off; off >>= 1) var += __shfl_xor(var, off, 32);
    var *= (1.f / DD);
    float inv = rsqrtf(var + 1e-5f);
    #pragma unroll
    for (int i = 0; i < 4; i++)
        out[base + i] = (v[i] - mu) * inv * s[lane * 4 + i] + b[lane * 4 + i];
}

// ---------------- masked mean pool over walk positions ----------------
__global__ __launch_bounds__(128)
void k_meanpool(const h16* __restrict__ x, const int* __restrict__ masks,
                h16* __restrict__ out) {
    int bw = blockIdx.x, d = threadIdx.x;
    const int* mp = masks + (size_t)bw * LL;
    float s = 0.f; float cnt = 0.f;
    for (int l = 0; l < LL; l++)
        if (mp[l]) { s += (float)x[((size_t)bw * LL + l) * DD + d]; cnt += 1.f; }
    out[(size_t)bw * DD + d] = (h16)(s / (cnt + 1e-8f));
}

// ---------------- node-id occurrence counts per batch ----------------
__global__ __launch_bounds__(128)
void k_count(const int* __restrict__ nodes, const int* __restrict__ masks,
             int* __restrict__ counts) {
    int b = blockIdx.x;
    __shared__ int c[LL];
    if (threadIdx.x < LL) c[threadIdx.x] = 0;
    __syncthreads();
    for (int i = threadIdx.x; i < WW * LL; i += 128)
        if (masks[(size_t)b * WW * LL + i])
            atomicAdd(&c[nodes[(size_t)b * WW * LL + i]], 1);
    __syncthreads();
    if (threadIdx.x < LL) counts[b * LL + threadIdx.x] = c[threadIdx.x];
}

// ---------------- co-occurrence matrix ----------------
__global__ __launch_bounds__(128)
void k_cooc(const int* __restrict__ nodes, const int* __restrict__ masks,
            const int* __restrict__ counts, float* __restrict__ cooc) {
    int bw = blockIdx.x;
    int b = bw >> 7, w = bw & 127;
    __shared__ int nw[LL], mw[LL];
    __shared__ float singl;
    __shared__ int lenw;
    if (threadIdx.x == 0) {
        int lw = 0; float sg = 0.f;
        for (int p = 0; p < LL; p++) {
            nw[p] = nodes[((size_t)b * WW + w) * LL + p];
            mw[p] = masks[((size_t)b * WW + w) * LL + p];
            lw += mw[p];
            if (mw[p] && counts[b * LL + nw[p]] == 1) sg += 1.f;
        }
        lenw = lw; singl = sg;
    }
    __syncthreads();
    int vv = threadIdx.x;
    const int* nv = nodes + ((size_t)b * WW + vv) * LL;
    const int* mv = masks + ((size_t)b * WW + vv) * LL;
    int lenv = 0;
    for (int q = 0; q < LL; q++) lenv += mv[q];
    float acc = 0.f;
    for (int p = 0; p < LL; p++) {
        if (!mw[p]) continue;
        int np = nw[p];
        for (int q = 0; q < LL; q++)
            if (mv[q] && nv[q] == np) {
                float dpq = (float)(p - q);
                acc += __expf(-dpq * dpq * 0.25f);   // sigma^2 = 4
            }
    }
    if (vv == w) acc -= singl;
    cooc[((size_t)b * WW + w) * WW + vv] = acc / ((float)lenw * (float)lenv + 1e-8f);
}

// ---------------- attention pooling over walks ----------------
__global__ __launch_bounds__(128)
void k_pool(const h16* __restrict__ t, const float* __restrict__ w2,
            const float* __restrict__ b2, const h16* __restrict__ y,
            const int* __restrict__ masks, h16* __restrict__ g) {
    int b = blockIdx.x, tid = threadIdx.x;
    __shared__ float sa[WW];
    __shared__ float red[2];
    const h16* tp = t + ((size_t)b * WW + tid) * DD;
    float s = b2[0];
    for (int d = 0; d < DD; d++) s += (float)tp[d] * w2[d];
    int len = 0;
    const int* mp = masks + ((size_t)b * WW + tid) * LL;
    for (int l = 0; l < LL; l++) len += mp[l];
    if (!len) s = -1e9f;
    sa[tid] = s;
    __syncthreads();
    if (tid == 0) {
        float mx = -1e30f;
        for (int i = 0; i < WW; i++) mx = fmaxf(mx, sa[i]);
        float dn = 0.f;
        for (int i = 0; i < WW; i++) dn += __expf(sa[i] - mx);
        red[0] = mx; red[1] = dn;
    }
    __syncthreads();
    float a = __expf(sa[tid] - red[0]) / red[1];
    sa[tid] = a;
    __syncthreads();
    float acc = 0.f;  // tid now indexes output dim
    for (int w = 0; w < WW; w++)
        acc += sa[w] * (float)y[((size_t)b * WW + w) * DD + tid];
    g[(size_t)b * DD + tid] = (h16)acc;
}

// =====================================================================
extern "C" void kernel_launch(void* const* d_in, const int* in_sizes, int n_in,
                              void* d_out, int out_size, void* d_ws, size_t ws_size,
                              hipStream_t stream) {
    const float* ne    = (const float*)d_in[0];
    const int*   nodes = (const int*)d_in[1];
    const int*   masks = (const int*)d_in[2];
    const float* te    = (const float*)d_in[3];
    auto P = [&](int i) { return (const float*)d_in[i]; };

    // ---- workspace layout ----
    char* wsb = (char*)d_ws;
    size_t off = 0;
    auto alloc = [&](size_t bytes) { void* p = wsb + off; off += (bytes + 255) & ~(size_t)255; return p; };
    h16* X    = (h16*)alloc(M1 * DD * 2);
    h16* Q    = (h16*)alloc(M1 * DD * 2);
    h16* Kb   = (h16*)alloc(M1 * DD * 2);
    h16* Vb   = (h16*)alloc(M1 * DD * 2);
    h16* O    = (h16*)alloc(M1 * DD * 2);
    h16* Hh   = (h16*)alloc(M1 * FFN * 2);
    h16* SUMA = (h16*)alloc(M2 * DD * 2);
    h16* SUMM = (h16*)alloc(M2 * DD * 2);
    h16* G    = (h16*)alloc((size_t)BB * DD * 2);
    h16* GO   = (h16*)alloc((size_t)BB * DD * 2);
    float* COOC = (float*)alloc((size_t)BB * WW * WW * 4);
    int* CNT    = (int*)alloc((size_t)BB * LL * 4);
    h16* wblob  = (h16*)alloc((size_t)700000 * 2);

    // ---- weight f16 conversion ----
    h16* wp = wblob;
    auto cvt = [&](const float* s, int n) {
        h16* d = wp; wp += n;
        k_cvt<<<(n + 255) / 256, 256, 0, stream>>>(s, d, n);
        return d;
    };
    struct LWt { h16 *Wq, *Wk, *Wv, *Wo, *W1, *W2; int base; };
    LWt lw[4];
    int bases[4] = {4, 20, 41, 57};   // intra0, intra1, inter0, inter1
    for (int t = 0; t < 4; t++) {
        int bs = bases[t];
        lw[t].base = bs;
        lw[t].Wq = cvt(P(bs + 0), DD * DD);
        lw[t].Wk = cvt(P(bs + 2), DD * DD);
        lw[t].Wv = cvt(P(bs + 4), DD * DD);
        lw[t].Wo = cvt(P(bs + 6), DD * DD);
        lw[t].W1 = cvt(P(bs + 8), DD * FFN);
        lw[t].W2 = cvt(P(bs + 10), FFN * DD);
    }
    h16* intraOutW = cvt(P(36), DD * DD);
    h16* poolW1    = cvt(P(75), DD * DD);
    h16* outW      = cvt(P(79), DD * DD);

    auto gemm = [&](const h16* A, const h16* Bm, const float* bias, h16* C,
                    int M, int N, int K, int act) {
        dim3 g((M + 127) / 128, (N + 63) / 64);
        k_gemm<<<g, 256, 0, stream>>>(A, Bm, bias, C, M, N, K, act);
    };
    auto ln = [&](const h16* x, const h16* r, const float* s, const float* b,
                  h16* o, int M) {
        k_ln<<<M, 32, 0, stream>>>(x, r, s, b, o, M);
    };

    // ---- embed + PE ----
    k_embed<<<(unsigned)((M1 * DD + 255) / 256), 256, 0, stream>>>(ne, te, X, (long)(M1 * DD));

    // ---- intra-walk encoder (2 layers) ----
    for (int t = 0; t < 2; t++) {
        int bs = lw[t].base;
        gemm(X, lw[t].Wq, P(bs + 1), Q,  (int)M1, DD, DD, 0);
        gemm(X, lw[t].Wk, P(bs + 3), Kb, (int)M1, DD, DD, 0);
        gemm(X, lw[t].Wv, P(bs + 5), Vb, (int)M1, DD, DD, 0);
        k_attn_intra<<<BB * WW * NH, 32, 0, stream>>>(Q, Kb, Vb, masks, O);
        gemm(O, lw[t].Wo, P(bs + 7), Q, (int)M1, DD, DD, 0);
        ln(X, Q, P(bs + 12), P(bs + 13), X, (int)M1);
        gemm(X, lw[t].W1, P(bs + 9), Hh, (int)M1, FFN, DD, 1);   // ReLU
        gemm(Hh, lw[t].W2, P(bs + 11), O, (int)M1, DD, FFN, 0);
        ln(X, O, P(bs + 14), P(bs + 15), X, (int)M1);
    }
    ln(X, nullptr, P(38), P(39), X, (int)M1);          // intra_norm

    // ---- walk summaries ----
    k_meanpool<<<(unsigned)M2, 128, 0, stream>>>(X, masks, SUMA);
    gemm(SUMA, intraOutW, P(37), SUMM, (int)M2, DD, DD, 0);
    ln(SUMM, nullptr, P(38), P(39), SUMM, (int)M2);

    // ---- co-occurrence bias ----
    k_count<<<BB, 128, 0, stream>>>(nodes, masks, CNT);
    k_cooc<<<BB * WW, 128, 0, stream>>>(nodes, masks, CNT, COOC);

    // ---- inter-walk encoder (2 layers) ----
    for (int t = 2; t < 4; t++) {
        int bs = lw[t].base;
        gemm(SUMM, lw[t].Wq, P(bs + 1), Q,  (int)M2, DD, DD, 0);
        gemm(SUMM, lw[t].Wk, P(bs + 3), Kb, (int)M2, DD, DD, 0);
        gemm(SUMM, lw[t].Wv, P(bs + 5), Vb, (int)M2, DD, DD, 0);
        k_attn_inter<<<BB * NH, 128, 0, stream>>>(Q, Kb, Vb, COOC, P(40), masks, O);
        gemm(O, lw[t].Wo, P(bs + 7), Q, (int)M2, DD, DD, 0);
        ln(SUMM, Q, P(bs + 12), P(bs + 13), SUMM, (int)M2);
        gemm(SUMM, lw[t].W1, P(bs + 9), Hh, (int)M2, FFN, DD, 1);
        gemm(Hh, lw[t].W2, P(bs + 11), O, (int)M2, DD, FFN, 0);
        ln(SUMM, O, P(bs + 14), P(bs + 15), SUMM, (int)M2);
    }
    ln(SUMM, nullptr, P(73), P(74), SUMM, (int)M2);    // inter_norm

    // ---- attention pooling over walks ----
    gemm(SUMM, poolW1, P(76), Q, (int)M2, DD, DD, 2);  // tanh
    k_pool<<<BB, 128, 0, stream>>>(Q, P(77), P(78), SUMM, masks, G);

    // ---- output projection + final LN (f32 out) ----
    gemm(G, outW, P(80), GO, BB, DD, DD, 0);
    k_ln_f32<<<BB, 32, 0, stream>>>(GO, P(81), P(82), (float*)d_out);
}
